// AqSolModel_54640573940143
// MI455X (gfx1250) — matrix operand order repaired and checked by reading, not compiled, not run on hardware
//
#include <hip/hip_runtime.h>
#include <stdint.h>

// ---------------------------------------------------------------------------
// CDNA5 (gfx1250) GNN forward: BN-fold -> scatter-add -> bf16 WMMA GEMMs
// Weights pre-transposed to [N][K] bf16; weight tiles staged into LDS via the
// Tensor Data Mover (tensor_load_to_lds) when available, else b128 copies.
// ---------------------------------------------------------------------------

typedef __attribute__((ext_vector_type(16))) __bf16 v16bf;
typedef __attribute__((ext_vector_type(8)))  float  v8f;
typedef __attribute__((ext_vector_type(4)))  unsigned int u32x4;
typedef __attribute__((ext_vector_type(4)))  int i32x4;
typedef __attribute__((ext_vector_type(8)))  int i32x8;

using u16 = unsigned short;
using u32 = unsigned int;

#if defined(__has_builtin)
#if __has_builtin(__builtin_amdgcn_tensor_load_to_lds) && \
    __has_builtin(__builtin_amdgcn_s_wait_tensorcnt)
#define USE_TDM 1
#endif
#endif
#ifndef USE_TDM
#define USE_TDM 0
#endif

static __device__ __forceinline__ u16 f2bf(float f) {
  u32 u = __float_as_uint(f);
  u32 r = u + 0x7FFFu + ((u >> 16) & 1u);   // round-to-nearest-even
  return (u16)(r >> 16);
}
static __device__ __forceinline__ u32 f2bf2(float x, float y) {
  return (u32)f2bf(x) | ((u32)f2bf(y) << 16);
}
static __device__ __forceinline__ float bf2f(u16 h) {
  return __uint_as_float(((u32)h) << 16);
}

// ---------------- small utility kernels ----------------

__global__ void k_zero(float* p, int n) {
  int i = blockIdx.x * 256 + threadIdx.x;
  if (i < n) p[i] = 0.0f;
}

// convert f32 W[K][N] -> bf16 Wt[N][K] (transposed, once per launch)
__global__ void k_cvt_bf16_t(const float* __restrict__ w, u16* __restrict__ wt,
                             int K, int N) {
  int i = blockIdx.x * 256 + threadIdx.x;
  if (i >= K * N) return;
  int k = i / N;
  int n = i % N;
  wt[(size_t)n * K + k] = f2bf(w[i]);
}

// fold eval-mode BN into per-channel scale/shift: y = x*s + t
__global__ void k_bn_fold(const float* g, const float* b, const float* m,
                          const float* v, float* s, float* t, int n) {
  int i = blockIdx.x * 256 + threadIdx.x;
  if (i >= n) return;
  float sc = g[i] * rsqrtf(v[i] + 1e-5f);
  s[i] = sc;
  t[i] = b[i] - m[i] * sc;
}

// agg = BN(h)  (self-loop term for the input layer), float4 vectorized
__global__ void k_bn_apply(const float* __restrict__ h, const float* __restrict__ s,
                           const float* __restrict__ t, float* __restrict__ agg,
                           int nNodes, int C) {
  int cg = C >> 2;
  int i = blockIdx.x * 256 + threadIdx.x;
  if (i >= nNodes * cg) return;
  int c4 = i % cg;
  float4 hv = ((const float4*)h)[i];
  float4 sv = ((const float4*)s)[c4];
  float4 tv = ((const float4*)t)[c4];
  float4 o;
  o.x = hv.x * sv.x + tv.x;
  o.y = hv.y * sv.y + tv.y;
  o.z = hv.z * sv.z + tv.z;
  o.w = hv.w * sv.w + tv.w;
  ((float4*)agg)[i] = o;
}

// agg[dst] += BN(h[src])   (input layer: BN applied on the fly)
__global__ void k_scatter_bn(const float* __restrict__ h, const float* __restrict__ s,
                             const float* __restrict__ t, const int* __restrict__ src,
                             const int* __restrict__ dst, float* __restrict__ agg,
                             int nEdges, int C) {
  int cg = C >> 2;
  int i = blockIdx.x * 256 + threadIdx.x;
  if (i >= nEdges * cg) return;
  int e  = i / cg;
  int c4 = i % cg;
  int sn = src[e], dn = dst[e];
  float4 hv = ((const float4*)(h + (size_t)sn * C))[c4];
  float4 sv = ((const float4*)s)[c4];
  float4 tv = ((const float4*)t)[c4];
  float* a = agg + (size_t)dn * C + c4 * 4;
  atomicAdd(a + 0, hv.x * sv.x + tv.x);
  atomicAdd(a + 1, hv.y * sv.y + tv.y);
  atomicAdd(a + 2, hv.z * sv.z + tv.z);
  atomicAdd(a + 3, hv.w * sv.w + tv.w);
}

// agg[dst] += y[src]   (conv layers: BN pre-applied by producing GEMM epilogue)
__global__ void k_scatter_raw(const float* __restrict__ y, const int* __restrict__ src,
                              const int* __restrict__ dst, float* __restrict__ agg,
                              int nEdges, int C) {
  int cg = C >> 2;
  int i = blockIdx.x * 256 + threadIdx.x;
  if (i >= nEdges * cg) return;
  int e  = i / cg;
  int c4 = i % cg;
  int sn = src[e], dn = dst[e];
  float4 hv = ((const float4*)(y + (size_t)sn * C))[c4];
  float* a = agg + (size_t)dn * C + c4 * 4;
  atomicAdd(a + 0, hv.x);
  atomicAdd(a + 1, hv.y);
  atomicAdd(a + 2, hv.z);
  atomicAdd(a + 3, hv.w);
}

// ---------------- WMMA bf16 GEMM ----------------
// C[M,N] = epilogue(A[M,K] @ W[K,N]);  Wt is bf16 pre-transposed [N][K].
// Block tile 128 x BNt, 8 waves (4x2): wave tile 32 x (BNt/2).
// K staged 64 deep (two WMMA K-steps per barrier).
// Weight tiles staged by the Tensor Data Mover with LDS padding matching PITCH.
// NEXT_BN: epilogue additionally applies the NEXT layer's input-BN and writes
// the result to both outF (gather source) and aggOut (self-loop seed).

#define BM 128
#define BK 64
#define PITCH 72   // padded LDS pitch (u16): 144B rows keep 16B alignment

union FragU { uint4 q[2]; v16bf v; };

template <int BNt, bool A_BF16, bool HAS_BN, bool RELU, bool OUT_BF16, bool NEXT_BN>
__global__ void __launch_bounds__(256) k_gemm(const void* __restrict__ Av,
                                              const u16*  __restrict__ Wt,
                                              const float* __restrict__ bias,
                                              const float* __restrict__ scale,
                                              const float* __restrict__ shift,
                                              const float* __restrict__ nscale,
                                              const float* __restrict__ nshift,
                                              float* __restrict__ aggOut,
                                              float* __restrict__ outF,
                                              u16*   __restrict__ outB,
                                              int M, int K, int N) {
  constexpr int TN = BNt / 32;   // N sub-tiles per wave (2 or 4)
  __shared__ __align__(16) u16 Al[BM * PITCH];
  __shared__ __align__(16) u16 Bl[BNt * PITCH];

  const int tid  = threadIdx.x;
  const int lane = tid & 31;
  const int wave = tid >> 5;
  const int wm = (wave & 3) * 32;          // wave M offset in block tile
  const int wn = (wave >> 2) * (BNt / 2);  // wave N offset
  const int m0 = blockIdx.x * BM;
  const int n0 = blockIdx.y * BNt;

  const int nl    = lane & 15;
  const int hi    = lane >> 4;   // lane half
  const int abase = hi * 8;      // A fragment K base (ISA 16-bit A layout)
  const int bbase = hi * 8;      // B fragment K base (ISA B layout)

  v8f acc[2][TN] = {};

  for (int k0 = 0; k0 < K; k0 += BK) {
    // ---- stage A tile (convert f32->bf16 if needed) ----
    if constexpr (A_BF16) {
      #pragma unroll
      for (int i = 0; i < (BM * BK) / (8 * 256); ++i) {   // 4 iters
        int e  = tid + i * 256;                 // uint4 units (8 u16 each)
        int r  = e >> 3;                        // 8 uint4 per row
        int c8 = (e & 7) * 8;
        uint4 q = make_uint4(0u, 0u, 0u, 0u);
        int gr = m0 + r;
        const u16* A = (const u16*)Av;
        if (gr < M) {
          q = *(const uint4*)(A + (size_t)gr * K + k0 + c8);
          if (k0 + BK < K)
            __builtin_prefetch(A + (size_t)gr * K + k0 + BK + c8, 0, 3);
        }
        *(uint4*)(Al + r * PITCH + c8) = q;
      }
    } else {
      #pragma unroll
      for (int i = 0; i < (BM * BK) / (4 * 256); ++i) {   // 8 iters
        int e  = tid + i * 256;                 // float4 units
        int r  = e >> 4;                        // 16 float4 per row
        int c4 = (e & 15) * 4;
        float4 f = make_float4(0.f, 0.f, 0.f, 0.f);
        int gr = m0 + r;
        const float* A = (const float*)Av;
        if (gr < M) {
          f = *(const float4*)(A + (size_t)gr * K + k0 + c4);
          if (k0 + BK < K)
            __builtin_prefetch(A + (size_t)gr * K + k0 + BK + c4, 0, 3);
        }
        uint2 pk;
        pk.x = f2bf2(f.x, f.y);
        pk.y = f2bf2(f.z, f.w);
        *(uint2*)(Al + r * PITCH + c4) = pk;
      }
    }
    // ---- stage W tile: Bl[n][k] = Wt[n0+n][k0+k] ----
#if USE_TDM
    // Tensor Data Mover: 2D tile BNt rows x 64 bf16, row stride K elements.
    // LDS pad: every 32 DWORDs (one 128B row) insert 4 DWORDs -> 144B pitch.
    if (wave == 0) {
      unsigned long long ga =
          (unsigned long long)(uintptr_t)(Wt + (size_t)n0 * K + k0);
      u32x4 g0;
      g0[0] = 1u;                                        // count=1 (valid D#)
      g0[1] = (u32)(uintptr_t)(&Bl[0]);                  // lds_addr (bytes)
      g0[2] = (u32)ga;                                   // global_addr[31:0]
      g0[3] = (u32)((ga >> 32) & 0x01FFFFFFu) | (2u << 30);  // addr[56:32]|type=2
      i32x8 g1;
      g1[0] = (1 << 16) | (1 << 20) | (4 << 22) | (3 << 25);
      //        ^2B elem   ^pad_en    ^every 32dw  ^pad +4dw
      g1[1] = (K & 0xFFFF) << 16;                        // tensor_dim0 lo16
      g1[2] = ((K >> 16) & 0xFFFF) | ((N & 0xFFFF) << 16);   // dim0 hi | dim1 lo
      g1[3] = ((N >> 16) & 0xFFFF) | (BK << 16);         // dim1 hi | tile_dim0=64
      g1[4] = BNt;                                       // tile_dim1 (tile_dim2=0)
      g1[5] = K;                                         // tensor_dim0_stride lo32
      g1[6] = 0;                                         // stride hi
      g1[7] = 0;
      i32x4 gz  = {0, 0, 0, 0};
      i32x8 gz8 = {0, 0, 0, 0, 0, 0, 0, 0};
      __builtin_amdgcn_tensor_load_to_lds(g0, g1, gz, gz, gz8, 0);
      __builtin_amdgcn_s_wait_tensorcnt(0);
    }
#else
    #pragma unroll
    for (int i = 0; i < (BNt * BK) / (8 * 256); ++i) {    // BNt/32 iters
      int e  = tid + i * 256;                  // uint4 units
      int r  = e >> 3;                         // 8 uint4 per row (BK=64)
      int c8 = (e & 7) * 8;
      uint4 q = *(const uint4*)(Wt + (size_t)(n0 + r) * K + k0 + c8);
      *(uint4*)(Bl + r * PITCH + c8) = q;
    }
#endif
    __syncthreads();

    // ---- two WMMA K-steps per stage ----
    #pragma unroll
    for (int kk = 0; kk < BK; kk += 32) {
      FragU a[2], b[TN];
      #pragma unroll
      for (int ti = 0; ti < 2; ++ti) {
        const u16* ap = Al + (wm + ti * 16 + nl) * PITCH + kk + abase;
        a[ti].q[0] = *(const uint4*)(ap);
        a[ti].q[1] = *(const uint4*)(ap + 16);
      }
      #pragma unroll
      for (int tj = 0; tj < TN; ++tj) {
        const u16* bp = Bl + (wn + tj * 16 + nl) * PITCH + kk + bbase;
        b[tj].q[0] = *(const uint4*)(bp);
        b[tj].q[1] = *(const uint4*)(bp + 16);
      }
      #pragma unroll
      for (int ti = 0; ti < 2; ++ti)
        #pragma unroll
        for (int tj = 0; tj < TN; ++tj)
          acc[ti][tj] = __builtin_amdgcn_wmma_f32_16x16x32_bf16(
              false, a[ti].v, false, b[tj].v, (short)0, acc[ti][tj], false, false);
    }
    __syncthreads();
  }

  // ---- fused epilogue: bias (+BN) (+ReLU) (+next-layer BN), stores ----
  const int rbase = hi * 8;   // C/D layout: M = r + 8*(lane>=16)
  #pragma unroll
  for (int ti = 0; ti < 2; ++ti) {
    #pragma unroll
    for (int tj = 0; tj < TN; ++tj) {
      int gn = n0 + wn + tj * 16 + nl;
      float bsv = bias[gn];
      float sc = 1.f, sh = 0.f;
      if constexpr (HAS_BN) { sc = scale[gn]; sh = shift[gn]; }
      float nsc = 1.f, nsh = 0.f;
      if constexpr (NEXT_BN) { nsc = nscale[gn]; nsh = nshift[gn]; }
      v8f c = acc[ti][tj];
      #pragma unroll
      for (int r = 0; r < 8; ++r) {
        int gm = m0 + wm + ti * 16 + rbase + r;
        if (gm >= M) continue;
        float val = c[r] + bsv;
        if constexpr (HAS_BN) val = val * sc + sh;
        if constexpr (RELU)   val = fmaxf(val, 0.f);
        size_t idx = (size_t)gm * N + gn;
        if constexpr (NEXT_BN) {
          float y = val * nsc + nsh;      // next layer's input BN
          outF[idx]   = y;                // gather source for edge scatter
          aggOut[idx] = y;                // self-loop seed
        } else if constexpr (OUT_BF16) {
          outB[idx] = f2bf(val);
        } else {
          outF[idx] = val;
        }
      }
    }
  }
}

// ---------------- pooling + head ----------------

__global__ void k_pool(const float* __restrict__ h, const int* __restrict__ batch,
                       float* __restrict__ sums, int nNodes, int C) {
  int cg = C >> 2;
  int i = blockIdx.x * 256 + threadIdx.x;
  if (i >= nNodes * cg) return;
  int nidx = i / cg, c4 = i % cg;
  int g = batch[nidx];
  float4 hv = ((const float4*)(h + (size_t)nidx * C))[c4];
  float* p = sums + (size_t)g * C + c4 * 4;
  atomicAdd(p + 0, hv.x);
  atomicAdd(p + 1, hv.y);
  atomicAdd(p + 2, hv.z);
  atomicAdd(p + 3, hv.w);
}

__global__ void k_cnt(const int* __restrict__ batch, float* __restrict__ cnt, int nNodes) {
  int i = blockIdx.x * 256 + threadIdx.x;
  if (i < nNodes) atomicAdd(&cnt[batch[i]], 1.0f);
}

__global__ void k_pool_div(const float* __restrict__ sums, const float* __restrict__ cnt,
                           u16* __restrict__ p, int nGraphs, int C) {
  int i = blockIdx.x * 256 + threadIdx.x;
  if (i >= nGraphs * C) return;
  float c = fmaxf(cnt[i / C], 1.0f);
  p[i] = f2bf(sums[i] / c);
}

// out[row] = dot(p[row], fw) + fb   (one wave32 per row)
__global__ void k_final(const u16* __restrict__ p, const float* __restrict__ fw,
                        const float* __restrict__ fb, float* __restrict__ out,
                        int nGraphs, int C) {
  int row  = blockIdx.x * 8 + (threadIdx.x >> 5);
  int lane = threadIdx.x & 31;
  if (row >= nGraphs) return;
  float s = 0.f;
  for (int i = lane; i < C; i += 32) s += bf2f(p[(size_t)row * C + i]) * fw[i];
  #pragma unroll
  for (int off = 16; off; off >>= 1) s += __shfl_xor(s, off, 32);
  if (lane == 0) out[row] = s + fb[0];
}

// ---------------------------------------------------------------------------

extern "C" void kernel_launch(void* const* d_in, const int* in_sizes, int n_in,
                              void* d_out, int out_size, void* d_ws, size_t ws_size,
                              hipStream_t stream) {
  (void)in_sizes; (void)n_in; (void)out_size; (void)ws_size;
  constexpr int NN = 50000, NE = 150000, NF = 128, H = 512, H1 = 320, NG = 2048;

  // ---- inputs (setup_inputs dict order) ----
  const float* x      = (const float*)d_in[0];
  const int*   ei     = (const int*)d_in[1];
  const int*   batch  = (const int*)d_in[2];
  const float* in_g1  = (const float*)d_in[3];
  const float* in_b1  = (const float*)d_in[4];
  const float* in_m1  = (const float*)d_in[5];
  const float* in_v1  = (const float*)d_in[6];
  const float* w1a    = (const float*)d_in[7];
  const float* b1a    = (const float*)d_in[8];
  const float* w1b    = (const float*)d_in[9];
  const float* b1b    = (const float*)d_in[10];
  const float* out_g1 = (const float*)d_in[11];
  const float* out_b1 = (const float*)d_in[12];
  const float* out_m1 = (const float*)d_in[13];
  const float* out_v1 = (const float*)d_in[14];
  const float* cin_g  = (const float*)d_in[15];
  const float* cin_b  = (const float*)d_in[16];
  const float* cin_m  = (const float*)d_in[17];
  const float* cin_v  = (const float*)d_in[18];
  const float* cwA    = (const float*)d_in[19];
  const float* cbA    = (const float*)d_in[20];
  const float* cwB    = (const float*)d_in[21];
  const float* cbB    = (const float*)d_in[22];
  const float* cout_g = (const float*)d_in[23];
  const float* cout_b = (const float*)d_in[24];
  const float* cout_m = (const float*)d_in[25];
  const float* cout_v = (const float*)d_in[26];
  const float* lw     = (const float*)d_in[27];
  const float* lb     = (const float*)d_in[28];
  const float* fw     = (const float*)d_in[29];
  const float* fb     = (const float*)d_in[30];
  const int* src = ei;
  const int* dst = ei + NE;

  // ---- workspace carve-up ----
  char* ws = (char*)d_ws;
  size_t off = 0;
  auto give = [&](size_t bytes) {
    size_t r = off;
    off = (off + bytes + 255) & ~(size_t)255;
    return r;
  };
  float* h    = (float*)(ws + give((size_t)NN * H * 4));
  float* agg  = (float*)(ws + give((size_t)NN * H * 4));
  u16*   tbuf = (u16*)  (ws + give((size_t)NN * H * 2));
  u16*   wbf  = (u16*)  (ws + give((size_t)3088384 * 2));
  float* bns  = (float*)(ws + give((size_t)9472 * 4));
  float* sums = (float*)(ws + give((size_t)NG * H * 4));
  float* cnt  = (float*)(ws + give((size_t)NG * 4));
  u16*   pA   = (u16*)  (ws + give((size_t)NG * H * 2));
  u16*   pB   = (u16*)  (ws + give((size_t)NG * H * 2));

  // transposed bf16 weights: Wt[N][K]
  u16* w1a_t = wbf;                  // [H1][NF]
  u16* w1b_t = w1a_t + NF * H1;      // [H][H1]
  u16* cwA_t = w1b_t + H1 * H;       // 4x [H][H]
  u16* cwB_t = cwA_t + 4 * H * H;    // 4x [H][H]
  u16* lw_t  = cwB_t + 4 * H * H;    // 3x [H][H]

  float* in_s = bns;
  float* in_t = in_s + NF;
  float* o1_s = in_t + NF;
  float* o1_t = o1_s + H;
  float* ci_s = o1_t + H;
  float* ci_t = ci_s + 4 * H;
  float* co_s = ci_t + 4 * H;
  float* co_t = co_s + 4 * H;

  auto blk1 = [](int n) { return dim3((n + 255) / 256); };
  dim3 thr(256);

  // ---- precompute: bf16 transposed weights + BN folds ----
  k_cvt_bf16_t<<<blk1(NF * H1), thr, 0, stream>>>(w1a, w1a_t, NF, H1);
  k_cvt_bf16_t<<<blk1(H1 * H), thr, 0, stream>>>(w1b, w1b_t, H1, H);
  for (int i = 0; i < 4; ++i) {
    k_cvt_bf16_t<<<blk1(H * H), thr, 0, stream>>>(cwA + (size_t)i * H * H,
                                                  cwA_t + (size_t)i * H * H, H, H);
    k_cvt_bf16_t<<<blk1(H * H), thr, 0, stream>>>(cwB + (size_t)i * H * H,
                                                  cwB_t + (size_t)i * H * H, H, H);
  }
  for (int i = 0; i < 3; ++i)
    k_cvt_bf16_t<<<blk1(H * H), thr, 0, stream>>>(lw + (size_t)i * H * H,
                                                  lw_t + (size_t)i * H * H, H, H);

  k_bn_fold<<<blk1(NF), thr, 0, stream>>>(in_g1, in_b1, in_m1, in_v1, in_s, in_t, NF);
  k_bn_fold<<<blk1(H), thr, 0, stream>>>(out_g1, out_b1, out_m1, out_v1, o1_s, o1_t, H);
  k_bn_fold<<<blk1(4 * H), thr, 0, stream>>>(cin_g, cin_b, cin_m, cin_v, ci_s, ci_t, 4 * H);
  k_bn_fold<<<blk1(4 * H), thr, 0, stream>>>(cout_g, cout_b, cout_m, cout_v, co_s, co_t, 4 * H);

  // ---- input conv: BN -> scatter -> GEMM(relu) -> GEMM(+BN,relu,+nextBN) ----
  k_bn_apply<<<blk1(NN * (NF / 4)), thr, 0, stream>>>(x, in_s, in_t, agg, NN, NF);
  k_scatter_bn<<<blk1(NE * (NF / 4)), thr, 0, stream>>>(x, in_s, in_t, src, dst, agg, NE, NF);
  {
    dim3 g1((NN + BM - 1) / BM, H1 / 64);
    k_gemm<64, false, false, true, true, false><<<g1, thr, 0, stream>>>(
        agg, w1a_t, b1a, nullptr, nullptr, nullptr, nullptr, nullptr,
        nullptr, tbuf, NN, NF, H1);
    // epilogue applies conv0's input BN and seeds agg for the next layer
    dim3 g2((NN + BM - 1) / BM, H / 128);
    k_gemm<128, true, true, true, false, true><<<g2, thr, 0, stream>>>(
        tbuf, w1b_t, b1b, o1_s, o1_t, ci_s, ci_t, agg,
        h, nullptr, NN, H1, H);
  }

  // ---- 4 conv layers ----
  dim3 gC((NN + BM - 1) / BM, H / 128);
  for (int i = 0; i < 4; ++i) {
    // h already holds bn_in_i(prev activation); agg already seeded with it
    k_scatter_raw<<<blk1(NE * (H / 4)), thr, 0, stream>>>(h, src, dst, agg, NE, H);
    k_gemm<128, false, false, true, true, false><<<gC, thr, 0, stream>>>(
        agg, cwA_t + (size_t)i * H * H, cbA + i * H,
        nullptr, nullptr, nullptr, nullptr, nullptr, nullptr, tbuf, NN, H, H);
    if (i < 3) {
      // fold next layer's input BN into this epilogue; write h=y and seed agg
      k_gemm<128, true, true, true, false, true><<<gC, thr, 0, stream>>>(
          tbuf, cwB_t + (size_t)i * H * H, cbB + i * H,
          co_s + i * H, co_t + i * H, ci_s + (i + 1) * H, ci_t + (i + 1) * H,
          agg, h, nullptr, NN, H, H);
    } else {
      // last conv: raw relu(bn_out(...)) output for pooling
      k_gemm<128, true, true, true, false, false><<<gC, thr, 0, stream>>>(
          tbuf, cwB_t + (size_t)i * H * H, cbB + i * H,
          co_s + i * H, co_t + i * H, nullptr, nullptr, nullptr,
          h, nullptr, NN, H, H);
    }
  }

  // ---- global mean pool ----
  k_zero<<<blk1(NG * H), thr, 0, stream>>>(sums, NG * H);
  k_zero<<<blk1(NG), thr, 0, stream>>>(cnt, NG);
  k_pool<<<blk1(NN * (H / 4)), thr, 0, stream>>>(h, batch, sums, NN, H);
  k_cnt<<<blk1(NN), thr, 0, stream>>>(batch, cnt, NN);
  k_pool_div<<<blk1(NG * H), thr, 0, stream>>>(sums, cnt, pA, NG, H);

  // ---- head MLP (3x 512x512 GEMM + final dot) ----
  u16* pin = pA;
  u16* pout = pB;
  dim3 gL((NG + BM - 1) / BM, H / 128);
  for (int i = 0; i < 3; ++i) {
    k_gemm<128, true, false, true, true, false><<<gL, thr, 0, stream>>>(
        pin, lw_t + (size_t)i * H * H, lb + i * H,
        nullptr, nullptr, nullptr, nullptr, nullptr, nullptr, pout, NG, H, H);
    u16* tmp = pin; pin = pout; pout = tmp;
  }
  k_final<<<dim3(NG / 8), thr, 0, stream>>>(pin, fw, fb, (float*)d_out, NG, H);
}
